// DGCNNDeepSets_6648609374926
// MI455X (gfx1250) — matrix operand order repaired and checked by reading, not compiled, not run on hardware
//
#include <hip/hip_runtime.h>
#include <hip/hip_bf16.h>
#include <math.h>

// ---------------------------------------------------------------------------
// DGCNN + DeepSets on MI455X (gfx1250), wave32, WMMA f32_16x16x32_f16.
//  - pooled = A@h + h: float4 copy + per-edge float atomics (memory bound part)
//  - conv GEMMs + phi GEMM on WMMA with zero-padded K/N so the inner loop has
//    NO bounds checks: A = 4x float4 loads + cvt, B = one 32B half16 load.
//  - weights pre-converted to f16 column-major [Npad][Kpad] once per call.
//  - msg buffer has row stride 128 (97 real + zero pad) so phi's K=97 GEMM
//    runs a clean K=128 loop over zeros.
// ---------------------------------------------------------------------------

typedef _Float16 half16 __attribute__((ext_vector_type(16)));
typedef float    float8 __attribute__((ext_vector_type(8)));

union AFrag { half16 v; _Float16 e[16]; };

#define MSGLD 128   // padded row stride of msg (real width 97)

__global__ void zero_f32_kernel(float* __restrict__ p, long n) {
  long t = (long)blockIdx.x * blockDim.x + threadIdx.x;
  if (t < n) p[t] = 0.0f;
}

// pooled[r, q*4..] = in[r, in_off + q*4..]  (float4 per thread)
__global__ void pooled_init_kernel(const float* __restrict__ in, int ldin, int in_off,
                                   int F, long nrows, float* __restrict__ pooled) {
  long t = (long)blockIdx.x * blockDim.x + threadIdx.x;
  int  per = F >> 2;
  long total = nrows * (long)per;
  if (t >= total) return;
  long r = t / per;
  int  q = (int)(t - r * (long)per) << 2;
  *(float4*)(pooled + r * (long)F + q) =
      *(const float4*)(in + r * (long)ldin + in_off + q);
}

// pooled[dst[e], :] += in[src[e], in_off : in_off+F]  -- 4 features per thread
__global__ void scatter_add_kernel(const float* __restrict__ in, int ldin, int in_off,
                                   const int* __restrict__ src, const int* __restrict__ dst,
                                   long E, int F, float* __restrict__ pooled) {
  long t = (long)blockIdx.x * blockDim.x + threadIdx.x;
  int  per = F >> 2;
  long total = E * (long)per;
  if (t >= total) return;
  long e = t / per;
  int  q = (int)(t - e * (long)per) << 2;
  int  s = src[e];
  int  d = dst[e];
  float4 v = *(const float4*)(in + (long)s * ldin + in_off + q);
  float* op = pooled + (long)d * F + q;
  atomicAdd(op + 0, v.x);
  atomicAdd(op + 1, v.y);
  atomicAdd(op + 2, v.z);
  atomicAdd(op + 3, v.w);
}

// Wh[n*Kpad + k] = f16( (k<Kdim && n<Nout) ? W[k*Nout + n] : 0 )
__global__ void convert_w_kernel(const float* __restrict__ W, int Kdim, int Nout,
                                 int Kpad, int Npad, _Float16* __restrict__ out) {
  int t = blockIdx.x * blockDim.x + threadIdx.x;
  if (t >= Kpad * Npad) return;
  int n = t / Kpad, k = t - n * Kpad;
  float v = (k < Kdim && n < Nout) ? W[(long)k * Nout + n] : 0.0f;
  out[t] = (_Float16)v;
}

// -------- WMMA fragment builders (per-lane, contiguous-K order shared with B) --------
// A (f16 16x32): lane half g owns row (lane&15); halves 0..15 -> K=g*16+i,
//                one run of 16 consecutive K (matches B).
__device__ inline half16 load_a_frag(const float* __restrict__ rowp, int kb, int g) {
  float4 a0 = *(const float4*)(rowp + kb + g * 16);
  float4 a1 = *(const float4*)(rowp + kb + g * 16 + 4);
  float4 a2 = *(const float4*)(rowp + kb + g * 16 + 8);
  float4 a3 = *(const float4*)(rowp + kb + g * 16 + 12);
  AFrag a;
  a.e[0]=(_Float16)a0.x; a.e[1]=(_Float16)a0.y; a.e[2]=(_Float16)a0.z; a.e[3]=(_Float16)a0.w;
  a.e[4]=(_Float16)a1.x; a.e[5]=(_Float16)a1.y; a.e[6]=(_Float16)a1.z; a.e[7]=(_Float16)a1.w;
  a.e[8]=(_Float16)a2.x; a.e[9]=(_Float16)a2.y; a.e[10]=(_Float16)a2.z; a.e[11]=(_Float16)a2.w;
  a.e[12]=(_Float16)a3.x; a.e[13]=(_Float16)a3.y; a.e[14]=(_Float16)a3.z; a.e[15]=(_Float16)a3.w;
  return a.v;
}

// One wave -> one 16x16 tile of out = tanh((in @ W + bias)/deg).
// in: [nrows, ldin] fp32, K = Kpad (multiple of 32, zero-padded data).
// Wh: f16 column-major [Npad][Kpad]. out at [row, out_off+col], stride ldout.
__global__ void __launch_bounds__(128)
linear_wmma_kernel(const float* __restrict__ in, int ldin, int Kpad,
                   const _Float16* __restrict__ Wh, const float* __restrict__ bias, int Nout,
                   const float* __restrict__ degs,
                   float* __restrict__ out, int ldout, int out_off, int nrows) {
  const int lane  = threadIdx.x & 31;
  const int wave  = threadIdx.x >> 5;
  const int mtile = blockIdx.x * (blockDim.x >> 5) + wave;
  const int row0  = mtile * 16;
  if (row0 >= nrows) return;                 // whole-wave uniform exit
  const int ncol0 = blockIdx.y * 16;
  const int g  = lane >> 4;
  const int ml = lane & 15;

  const float*    arow = in + (long)(row0 + ml) * ldin;
  const _Float16* bcol_p = Wh + (long)(ncol0 + ml) * Kpad;

  float8 c = {};
  for (int kb = 0; kb < Kpad; kb += 32) {
    half16 a = load_a_frag(arow, kb, g);
    half16 b = *(const half16*)(bcol_p + kb + g * 16);   // K = g*16 + 0..15
    c = __builtin_amdgcn_wmma_f32_16x16x32_f16(false, a, false, b,
                                               (short)0, c, false, false);
  }

  // C/D layout: VGPR r -> row = row0 + g*8 + r, col = ncol0 + ml
  const int col = ncol0 + ml;
  if (col < Nout) {
    float bc = bias[col];
#pragma unroll
    for (int r = 0; r < 8; ++r) {
      int row = row0 + g * 8 + r;
      float v = (c[r] + bc) / degs[row];
      out[(long)row * ldout + out_off + col] = tanhf(v);
    }
  }
}

// One wave -> one 16x16 tile of relu(msg @ phi_W + phi_b), reduced over the
// tile's 16 rows in-register, one atomicAdd per column into summed[graph,col].
__global__ void __launch_bounds__(128)
phi_wmma_kernel(const float* __restrict__ msg, int ldin, int Kpad,
                const _Float16* __restrict__ Wh, const float* __restrict__ phib, int Nout,
                int npg, float* __restrict__ summed, int nrows) {
  const int lane  = threadIdx.x & 31;
  const int wave  = threadIdx.x >> 5;
  const int mtile = blockIdx.x * (blockDim.x >> 5) + wave;
  const int row0  = mtile * 16;
  if (row0 >= nrows) return;
  const int ncol0 = blockIdx.y * 16;
  const int g  = lane >> 4;
  const int ml = lane & 15;

  const float*    arow = msg + (long)(row0 + ml) * ldin;
  const _Float16* bcol_p = Wh + (long)(ncol0 + ml) * Kpad;

  float8 c = {};
  for (int kb = 0; kb < Kpad; kb += 32) {
    half16 a = load_a_frag(arow, kb, g);
    half16 b = *(const half16*)(bcol_p + kb + g * 16);
    c = __builtin_amdgcn_wmma_f32_16x16x32_f16(false, a, false, b,
                                               (short)0, c, false, false);
  }

  const int col = ncol0 + ml;
  float bc = (col < Nout) ? phib[col] : 0.0f;
  float partial = 0.0f;
#pragma unroll
  for (int r = 0; r < 8; ++r) {
    float v = c[r] + bc;
    partial += (v > 0.0f) ? v : 0.0f;
  }
  partial += __shfl_xor(partial, 16, 32);    // fold rows 8..15 onto 0..7 half
  if (g == 0 && col < Nout) {
    int graph = row0 / npg;                  // npg % 16 == 0 -> tile in one graph
    atomicAdd(&summed[(long)graph * Nout + col], partial);
  }
}

// out[b,o] = sum_k summed[b,k]*rhoW[k,o] + rhob[o]   (32x30x32, trivial)
__global__ void rho_kernel(const float* __restrict__ summed, const float* __restrict__ rhoW,
                           const float* __restrict__ rhob, int Bq, int Kq, int Oq,
                           float* __restrict__ out) {
  int t = blockIdx.x * blockDim.x + threadIdx.x;
  if (t >= Bq * Oq) return;
  int b = t / Oq, o = t - b * Oq;
  float acc = rhob[o];
  for (int k = 0; k < Kq; ++k)
    acc += summed[(long)b * Kq + k] * rhoW[(long)k * Oq + o];
  out[t] = acc;
}

static inline long cdiv(long a, long b) { return (a + b - 1) / b; }
static inline int pad_up(int x, int m) { return (x + m - 1) / m * m; }

extern "C" void kernel_launch(void* const* d_in, const int* in_sizes, int n_in,
                              void* d_out, int out_size, void* d_ws, size_t ws_size,
                              hipStream_t stream) {
  const float* node_feat = (const float*)d_in[0];
  const float* node_degs = (const float*)d_in[1];
  const int*   edge      = (const int*)d_in[2];
  const float* W[4]  = {(const float*)d_in[3], (const float*)d_in[5],
                        (const float*)d_in[7], (const float*)d_in[9]};
  const float* bv[4] = {(const float*)d_in[4], (const float*)d_in[6],
                        (const float*)d_in[8], (const float*)d_in[10]};
  const float* phiW = (const float*)d_in[11];
  const float* phib = (const float*)d_in[12];
  const float* rhoW = (const float*)d_in[13];
  const float* rhob = (const float*)d_in[14];

  const int  N    = in_sizes[1];            // node_degs has N elements
  const int  F_IN = in_sizes[0] / N;        // 128
  const long E    = in_sizes[2] / 2;        // 1M
  const int  Wout[4] = {in_sizes[4], in_sizes[6], in_sizes[8], in_sizes[10]}; // 32,32,32,1
  const int  K_PHI = in_sizes[12];          // 30
  const int  OUT   = in_sizes[14];          // 32
  const int  TOTALW = Wout[0] + Wout[1] + Wout[2] + Wout[3];                  // 97
  const int  NPG = 2048;
  const int  Bq  = N / NPG;                 // 32

  const int* src = edge;
  const int* dst = edge + E;

  // ---- workspace layout (fp32 region, then f16 weight region) ----
  float* pooled = (float*)d_ws;                       // N * F_IN
  float* msg    = pooled + (long)N * F_IN;            // N * MSGLD (padded)
  float* summed = msg + (long)N * MSGLD;              // Bq * K_PHI
  _Float16* whalf = (_Float16*)(summed + (long)Bq * K_PHI);

  int Kpad[5], Npad[5];                               // 4 conv layers + phi
  int Kdim[5] = {F_IN, Wout[0], Wout[1], Wout[2], TOTALW};
  int Ndim[5] = {Wout[0], Wout[1], Wout[2], Wout[3], K_PHI};
  _Float16* Wh[5];
  {
    _Float16* p = whalf;
    for (int l = 0; l < 5; ++l) {
      Kpad[l] = pad_up(Kdim[l], 32);
      Npad[l] = pad_up(Ndim[l], 16);
      Wh[l] = p;
      p += (long)Kpad[l] * Npad[l];
    }
  }

  int outoff[4];
  outoff[0] = 0;
  for (int l = 1; l < 4; ++l) outoff[l] = outoff[l - 1] + Wout[l - 1];

  const int WPB = 4;
  const int TPB = WPB * 32;

  // zero msg (incl. pad cols 97..127) and summed in one shot (adjacent)
  long zn = (long)N * MSGLD + (long)Bq * K_PHI;
  zero_f32_kernel<<<cdiv(zn, 256), 256, 0, stream>>>(msg, zn);

  // convert all weights to padded f16 column-major
  for (int l = 0; l < 5; ++l) {
    const float* Wsrc = (l < 4) ? W[l] : phiW;
    int n = Kpad[l] * Npad[l];
    convert_w_kernel<<<cdiv(n, 256), 256, 0, stream>>>(Wsrc, Kdim[l], Ndim[l],
                                                       Kpad[l], Npad[l], Wh[l]);
  }

  for (int l = 0; l < 4; ++l) {
    const float* in  = (l == 0) ? node_feat : msg;
    int ldin   = (l == 0) ? F_IN : MSGLD;
    int in_off = (l == 0) ? 0 : outoff[l - 1];
    int Fl     = Kdim[l];                             // input width of layer l

    long tot = (long)N * (Fl >> 2);
    pooled_init_kernel<<<cdiv(tot, 256), 256, 0, stream>>>(in, ldin, in_off, Fl, N, pooled);

    long stot = E * (long)(Fl >> 2);
    scatter_add_kernel<<<cdiv(stot, 256), 256, 0, stream>>>(in, ldin, in_off, src, dst,
                                                            E, Fl, pooled);

    dim3 grid((unsigned)cdiv(N / 16, WPB), (unsigned)(Npad[l] / 16));
    linear_wmma_kernel<<<grid, TPB, 0, stream>>>(pooled, Fl, Kpad[l],
                                                 Wh[l], bv[l], Wout[l],
                                                 node_degs, msg, MSGLD, outoff[l], N);
  }

  dim3 pgrid((unsigned)cdiv(N / 16, WPB), (unsigned)(Npad[4] / 16));
  phi_wmma_kernel<<<pgrid, TPB, 0, stream>>>(msg, MSGLD, Kpad[4], Wh[4], phib, K_PHI,
                                             NPG, summed, N);

  rho_kernel<<<cdiv((long)Bq * OUT, 256), 256, 0, stream>>>(summed, rhoW, rhob,
                                                            Bq, K_PHI, OUT, (float*)d_out);
}
